// CAM_Module_17841294147821
// MI455X (gfx1250) — compile-verified, matrix operand order checked by reading, not run on hardware
//
#include <hip/hip_runtime.h>
#include <hip/hip_bf16.h>
#include <stdint.h>

// CAM module: energy = Xf·Xf^T ; att = softmax(energy) ; out = gamma*(att·Xf) + x
// B=16, C=512, H=W=64 -> N=4096. bf16 WMMA GEMMs, f32 accumulation.
// B panel staged via async global->LDS (ASYNCcnt), double buffered with
// explicit pointer swap + unroll(1) to keep one accumulator set in place.

typedef __attribute__((ext_vector_type(16))) __bf16 v16bf;
typedef __attribute__((ext_vector_type(8)))  __bf16 v8bf;
typedef __attribute__((ext_vector_type(8)))  float  v8f;

// Types matching the async-to-LDS builtin signature: (int4 AS1*, int4 AS3*, imm, imm)
typedef __attribute__((__vector_size__(16))) int i32x4;
typedef __attribute__((address_space(1))) i32x4 gbl_i32x4;
typedef __attribute__((address_space(3))) i32x4 lds_i32x4;

#define BATCH 16
#define CCH   512
#define HW    4096

#if defined(__has_builtin)
#if __has_builtin(__builtin_amdgcn_global_load_async_to_lds_b128)
#define HAVE_ASYNC_LDS 1
#endif
#endif
#ifndef HAVE_ASYNC_LDS
#define HAVE_ASYNC_LDS 0
#endif

__device__ __forceinline__ unsigned short f32_to_bf16(float f) {
    unsigned int u = __float_as_uint(f);
    u += 0x7FFFu + ((u >> 16) & 1u);          // round-to-nearest-even
    return (unsigned short)(u >> 16);
}

// 16-element bf16 fragment: two 16B chunks at +0 and +16 elems.
// Works for both global (global_load_b128) and LDS (ds_load_b128) pointers.
__device__ __forceinline__ v16bf load_frag(const unsigned short* p) {
    union { v16bf v; v8bf h[2]; } u;
    u.h[0] = *reinterpret_cast<const v8bf*>(p);
    u.h[1] = *reinterpret_cast<const v8bf*>(p + 16);
    return u.v;
}

#define WMMA_BF16(A, Bm, Cm) \
    __builtin_amdgcn_wmma_f32_16x16x32_bf16(false, (A), false, (Bm), (short)0, (Cm), false, false)

// Stage a 64-row x 32-col bf16 panel (row stride = stride elems) into LDS.
// 128 threads: thread t covers row t>>1, 16-elem half t&1 (two b128s).
__device__ __forceinline__ void stage_panel(const unsigned short* __restrict__ gbase,
                                            size_t stride,
                                            unsigned short* lds_dst,
                                            int tid) {
    const int row  = tid >> 1;
    const int half = tid & 1;
    const unsigned short* gp = gbase + (size_t)row * stride + half * 16;
    unsigned short* lp = lds_dst + row * 32 + half * 16;
#if HAVE_ASYNC_LDS
    __builtin_amdgcn_global_load_async_to_lds_b128(
        (gbl_i32x4*)(unsigned short*)(uintptr_t)gp,
        (lds_i32x4*)lp, 0, 0);
    __builtin_amdgcn_global_load_async_to_lds_b128(
        (gbl_i32x4*)(unsigned short*)(uintptr_t)(gp + 8),
        (lds_i32x4*)(lp + 8), 0, 0);
#else
    *reinterpret_cast<v8bf*>(lp)     = *reinterpret_cast<const v8bf*>(gp);
    *reinterpret_cast<v8bf*>(lp + 8) = *reinterpret_cast<const v8bf*>(gp + 8);
#endif
}

__device__ __forceinline__ void wait_async_lds() {
#if HAVE_ASYNC_LDS
    asm volatile("s_wait_asynccnt 0x0" ::: "memory");
#endif
}

// ---------------- Kernel 1: f32 -> bf16, row-major + transposed copies
__global__ void cam_convert(const float* __restrict__ x,
                            unsigned short* __restrict__ xh,
                            unsigned short* __restrict__ xhT) {
    size_t idx = (size_t)blockIdx.x * blockDim.x + threadIdx.x;
    int n = (int)(idx % HW);
    size_t rc = idx / HW;
    int c = (int)(rc % CCH);
    int b = (int)(rc / CCH);
    unsigned short h = f32_to_bf16(x[idx]);
    xh[idx] = h;
    xhT[((size_t)b * HW + n) * CCH + c] = h;
}

// ---------------- Kernel 2: energy[b] = xh[b] (512xN) * xh[b]^T
// 128-thread block = 4 waves; block tile 128(M) x 64(N); wave tile 32(M) x 64(N).
__global__ void __launch_bounds__(128)
cam_energy_gemm(const unsigned short* __restrict__ xh,
                float* __restrict__ energy) {
    __shared__ unsigned short tile[2][64 * 32];   // 2 x 4KB double buffer

    const int tid  = threadIdx.x;
    const int wave = tid >> 5;
    const int lane = tid & 31;
    const int mr   = lane & 15;
    const int koff = (lane >> 4) * 8;
    const int mhi  = (lane >> 4) * 8;

    const int m0 = blockIdx.x * 128 + wave * 32;  // this wave: rows m0..m0+31
    const int j0 = blockIdx.y * 64;
    const int b  = blockIdx.z;

    const unsigned short* arow0 = xh + ((size_t)(b * CCH + m0 +      mr)) * HW + koff;
    const unsigned short* arow1 = xh + ((size_t)(b * CCH + m0 + 16 + mr)) * HW + koff;
    const unsigned short* bbase = xh + ((size_t)(b * CCH + j0)) * HW;   // 64 rows, stride HW

    unsigned short* cur = tile[0];
    unsigned short* nxt = tile[1];
    stage_panel(bbase, HW, cur, tid);

    v8f acc00 = {}, acc01 = {}, acc02 = {}, acc03 = {};
    v8f acc10 = {}, acc11 = {}, acc12 = {}, acc13 = {};
    const int ksteps = HW / 32;
#pragma unroll 1
    for (int i = 0; i < ksteps; ++i) {
        const int kb = i * 32;
        wait_async_lds();
        __syncthreads();                              // panel i visible to all waves
        if (i + 1 < ksteps)
            stage_panel(bbase + kb + 32, HW, nxt, tid);

        __builtin_prefetch(arow0 + kb + 256, 0, 1);
        __builtin_prefetch(arow1 + kb + 256, 0, 1);
        v16bf a0 = load_frag(arow0 + kb);
        v16bf a1 = load_frag(arow1 + kb);
        v16bf b0 = load_frag(cur + ( 0 + mr) * 32 + koff);
        v16bf b1 = load_frag(cur + (16 + mr) * 32 + koff);
        v16bf b2 = load_frag(cur + (32 + mr) * 32 + koff);
        v16bf b3 = load_frag(cur + (48 + mr) * 32 + koff);
        acc00 = WMMA_BF16(a0, b0, acc00);
        acc01 = WMMA_BF16(a0, b1, acc01);
        acc02 = WMMA_BF16(a0, b2, acc02);
        acc03 = WMMA_BF16(a0, b3, acc03);
        acc10 = WMMA_BF16(a1, b0, acc10);
        acc11 = WMMA_BF16(a1, b1, acc11);
        acc12 = WMMA_BF16(a1, b2, acc12);
        acc13 = WMMA_BF16(a1, b3, acc13);
        __syncthreads();                              // all reads done before next overwrite

        unsigned short* t = cur; cur = nxt; nxt = t;
    }

    for (int r = 0; r < 8; ++r) {
        int M0 = m0 + r + mhi;
        int M1 = m0 + 16 + r + mhi;
        float* erow0 = energy + ((size_t)(b * CCH + M0)) * CCH;
        float* erow1 = energy + ((size_t)(b * CCH + M1)) * CCH;
        erow0[j0 +  0 + mr] = acc00[r];
        erow0[j0 + 16 + mr] = acc01[r];
        erow0[j0 + 32 + mr] = acc02[r];
        erow0[j0 + 48 + mr] = acc03[r];
        erow1[j0 +  0 + mr] = acc10[r];
        erow1[j0 + 16 + mr] = acc11[r];
        erow1[j0 + 32 + mr] = acc12[r];
        erow1[j0 + 48 + mr] = acc13[r];
    }
}

// ---------------- Kernel 3: row softmax (512 wide), f32 in -> bf16 out
__global__ void cam_softmax(const float* __restrict__ energy,
                            unsigned short* __restrict__ att) {
    __shared__ float red[256];
    const int row = blockIdx.x;
    const int t   = threadIdx.x;
    const float* e = energy + (size_t)row * CCH;
    float v0 = e[t], v1 = e[t + 256];

    red[t] = fmaxf(v0, v1);
    __syncthreads();
    for (int s = 128; s > 0; s >>= 1) {
        if (t < s) red[t] = fmaxf(red[t], red[t + s]);
        __syncthreads();
    }
    float rowmax = red[0];
    __syncthreads();

    float e0 = __expf(v0 - rowmax);
    float e1 = __expf(v1 - rowmax);
    red[t] = e0 + e1;
    __syncthreads();
    for (int s = 128; s > 0; s >>= 1) {
        if (t < s) red[t] += red[t + s];
        __syncthreads();
    }
    float inv = 1.0f / red[0];

    unsigned short* arow = att + (size_t)row * CCH;
    arow[t]       = f32_to_bf16(e0 * inv);
    arow[t + 256] = f32_to_bf16(e1 * inv);
}

// ---------------- Kernel 4: out = gamma * (att · Xf) + x
// B-operand columns = xhT rows (contiguous); same LDS-staged structure.
__global__ void __launch_bounds__(128)
cam_out_gemm(const unsigned short* __restrict__ att,
             const unsigned short* __restrict__ xhT,
             const float* __restrict__ x,
             const float* __restrict__ gamma,
             float* __restrict__ out) {
    __shared__ unsigned short tile[2][64 * 32];

    const int tid  = threadIdx.x;
    const int wave = tid >> 5;
    const int lane = tid & 31;
    const int mr   = lane & 15;
    const int koff = (lane >> 4) * 8;
    const int mhi  = (lane >> 4) * 8;

    const int m0 = blockIdx.x * 128 + wave * 32;
    const int n0 = blockIdx.y * 64;
    const int b  = blockIdx.z;

    const unsigned short* arow0 = att + ((size_t)(b * CCH + m0 +      mr)) * CCH + koff;
    const unsigned short* arow1 = att + ((size_t)(b * CCH + m0 + 16 + mr)) * CCH + koff;
    const unsigned short* bbase = xhT + ((size_t)(b * HW + n0)) * CCH;  // 64 rows, stride CCH

    unsigned short* cur = tile[0];
    unsigned short* nxt = tile[1];
    stage_panel(bbase, CCH, cur, tid);

    v8f acc00 = {}, acc01 = {}, acc02 = {}, acc03 = {};
    v8f acc10 = {}, acc11 = {}, acc12 = {}, acc13 = {};
    const int ksteps = CCH / 32;
#pragma unroll 1
    for (int i = 0; i < ksteps; ++i) {
        const int kb = i * 32;
        wait_async_lds();
        __syncthreads();
        if (i + 1 < ksteps)
            stage_panel(bbase + kb + 32, CCH, nxt, tid);

        __builtin_prefetch(arow0 + kb + 128, 0, 1);
        __builtin_prefetch(arow1 + kb + 128, 0, 1);
        v16bf a0 = load_frag(arow0 + kb);
        v16bf a1 = load_frag(arow1 + kb);
        v16bf b0 = load_frag(cur + ( 0 + mr) * 32 + koff);
        v16bf b1 = load_frag(cur + (16 + mr) * 32 + koff);
        v16bf b2 = load_frag(cur + (32 + mr) * 32 + koff);
        v16bf b3 = load_frag(cur + (48 + mr) * 32 + koff);
        acc00 = WMMA_BF16(a0, b0, acc00);
        acc01 = WMMA_BF16(a0, b1, acc01);
        acc02 = WMMA_BF16(a0, b2, acc02);
        acc03 = WMMA_BF16(a0, b3, acc03);
        acc10 = WMMA_BF16(a1, b0, acc10);
        acc11 = WMMA_BF16(a1, b1, acc11);
        acc12 = WMMA_BF16(a1, b2, acc12);
        acc13 = WMMA_BF16(a1, b3, acc13);
        __syncthreads();

        unsigned short* t = cur; cur = nxt; nxt = t;
    }

    const float g = *gamma;
    for (int r = 0; r < 8; ++r) {
        int M0 = m0 + r + mhi;
        int M1 = m0 + 16 + r + mhi;
        size_t base0 = ((size_t)(b * CCH + M0)) * HW + n0 + mr;
        size_t base1 = ((size_t)(b * CCH + M1)) * HW + n0 + mr;
        out[base0 +  0] = g * acc00[r] + x[base0 +  0];
        out[base0 + 16] = g * acc01[r] + x[base0 + 16];
        out[base0 + 32] = g * acc02[r] + x[base0 + 32];
        out[base0 + 48] = g * acc03[r] + x[base0 + 48];
        out[base1 +  0] = g * acc10[r] + x[base1 +  0];
        out[base1 + 16] = g * acc11[r] + x[base1 + 16];
        out[base1 + 32] = g * acc12[r] + x[base1 + 32];
        out[base1 + 48] = g * acc13[r] + x[base1 + 48];
    }
}

extern "C" void kernel_launch(void* const* d_in, const int* in_sizes, int n_in,
                              void* d_out, int out_size, void* d_ws, size_t ws_size,
                              hipStream_t stream) {
    const float* x     = (const float*)d_in[0];
    const float* gamma = (const float*)d_in[1];
    float* out = (float*)d_out;

    char* ws = (char*)d_ws;
    const size_t xh_bytes = (size_t)BATCH * CCH * HW * sizeof(unsigned short);   // 64 MB
    const size_t en_bytes = (size_t)BATCH * CCH * CCH * sizeof(float);           // 16 MB
    const size_t at_bytes = (size_t)BATCH * CCH * CCH * sizeof(unsigned short);  //  8 MB
    unsigned short* xh     = (unsigned short*)(ws);
    unsigned short* xhT    = (unsigned short*)(ws + xh_bytes);
    float*          energy = (float*)(ws + 2 * xh_bytes);
    unsigned short* att    = (unsigned short*)(ws + 2 * xh_bytes + en_bytes);

    if (ws_size < 2 * xh_bytes + en_bytes + at_bytes)
        return;  // workspace too small; deterministic no-op

    {   // 1) f32 -> bf16 (row-major + transposed)
        size_t total = (size_t)BATCH * CCH * HW;
        cam_convert<<<(unsigned)(total / 256), 256, 0, stream>>>(x, xh, xhT);
    }
    // 2) energy = Xf * Xf^T
    cam_energy_gemm<<<dim3(CCH / 128, CCH / 64, BATCH), 128, 0, stream>>>(xh, energy);
    // 3) attention = softmax(energy)
    cam_softmax<<<BATCH * CCH, 256, 0, stream>>>(energy, att);
    // 4) out = gamma * (attention * Xf) + x
    cam_out_gemm<<<dim3(CCH / 128, HW / 64, BATCH), 128, 0, stream>>>(att, xhT, x, gamma, out);
}
